// CrystalformerMultiheadAttentionCUDA_61942018342984
// MI455X (gfx1250) — matrix operand order repaired and checked by reading, not compiled, not run on hardware
//
#include <hip/hip_runtime.h>
#include <hip/hip_bf16.h>

typedef __attribute__((ext_vector_type(16))) _Float16 v16h;
typedef __attribute__((ext_vector_type(8)))  _Float16 v8h;
typedef __attribute__((ext_vector_type(4)))  _Float16 v4h;
typedef __attribute__((ext_vector_type(8)))  float    v8f;

#define G_CRY 64
#define NA    64
#define NH    8
#define DH    64

// One block per (crystal, head). 128 threads = 4 waves (wave32).
__global__ __launch_bounds__(128)
void crystal_attn_kernel(const float* __restrict__ q,
                         const float* __restrict__ k,
                         const float* __restrict__ v,
                         const float* __restrict__ aw,      // (M, H) additive bias
                         const float* __restrict__ values,  // (M, H, DH) fused value term
                         float* __restrict__ out)           // (N, H, DH)
{
    __shared__ __attribute__((aligned(64))) _Float16 Qs[NA * DH];  // 8 KB, row-major f16
    __shared__ __attribute__((aligned(64))) _Float16 Ks[NA * DH];  // 8 KB, row-major f16
    __shared__ __attribute__((aligned(64))) float    Vs[NA * DH];  // 16 KB, row-major f32
    __shared__ __attribute__((aligned(64))) float    Ss[NA * NA];  // 16 KB, logits -> softmax weights

    const int c    = blockIdx.x >> 3;   // crystal
    const int h    = blockIdx.x & 7;    // head
    const int t    = threadIdx.x;
    const int lane = t & 31;
    const int w    = t >> 5;            // wave id 0..3

    // ---------------- Phase A: stage Q,K (f16) and V (f32) into LDS ----------
    // 64 rows x 16 float4-chunks = 1024 chunks, 8 per thread, coalesced.
    for (int idx = t; idx < NA * (DH / 4); idx += 128) {
        int row = idx >> 4;
        int c4  = (idx & 15) * 4;
        size_t gofs = ((size_t)((c * NA + row) * NH + h)) * DH + c4;
        float4 qv = *(const float4*)(q + gofs);
        float4 kv = *(const float4*)(k + gofs);
        float4 vv = *(const float4*)(v + gofs);
        v4h qh = { (_Float16)qv.x, (_Float16)qv.y, (_Float16)qv.z, (_Float16)qv.w };
        v4h kh = { (_Float16)kv.x, (_Float16)kv.y, (_Float16)kv.z, (_Float16)kv.w };
        *(v4h*)&Qs[row * DH + c4] = qh;
        *(v4h*)&Ks[row * DH + c4] = kh;
        *(float4*)&Vs[row * DH + c4] = vv;
    }
    __syncthreads();

    // ---------------- Phase B: logits = (Q K^T) * 1/8 + bias via WMMA --------
    // Wave w owns rows [16w, 16w+16). 4 column tiles x 2 k-steps of
    // v_wmma_f32_16x16x32_f16. A/B fragments use contiguous k = 16*half + s.
    {
        const int half = lane >> 4;      // 0: lanes 0-15, 1: lanes 16-31
        const int l15  = lane & 15;
        const int arow = w * 16 + l15;   // A-matrix row for this lane

        for (int nt = 0; nt < 4; ++nt) {
            v8f acc = {};
#pragma unroll
            for (int ks = 0; ks < 2; ++ks) {
                // A fragment: K chunks {kb..kb+7} and {kb+8..kb+15}
                int kb = ks * 32 + half * 16;
                v8h lo = *(const v8h*)&Qs[arow * DH + kb];
                v8h hi = *(const v8h*)&Qs[arow * DH + kb + 8];
                v16h a = __builtin_shufflevector(lo, hi,
                         0,1,2,3,4,5,6,7,8,9,10,11,12,13,14,15);
                // B fragment: column nt*16+l15 of K^T = row of K, contiguous 16 halfs
                int bcol = nt * 16 + l15;
                int bkb  = ks * 32 + half * 16;
                v16h b = *(const v16h*)&Ks[bcol * DH + bkb];
                acc = __builtin_amdgcn_wmma_f32_16x16x32_f16(
                          false, a, false, b, (short)0, acc, false, false);
            }
            // C/D layout: VGPR r -> row r + 8*half; col = nt*16 + l15
            const int colg = nt * 16 + l15;
#pragma unroll
            for (int r = 0; r < 8; ++r) {
                int rowl = w * 16 + half * 8 + r;
                size_t e = (size_t)c * (NA * NA) + (size_t)rowl * NA + colg;
                float bias = aw[e * NH + h];
                Ss[rowl * NA + colg] = acc[r] * 0.125f + bias;
            }
        }
    }
    __syncthreads();

    // ---------------- Phase C: row softmax in LDS ----------------------------
    if (t < NA) {
        float m = -3.402823466e38f;
        for (int j = 0; j < NA; ++j) m = fmaxf(m, Ss[t * NA + j]);
        float s = 0.0f;
        for (int j = 0; j < NA; ++j) {
            float e = __expf(Ss[t * NA + j] - m);
            Ss[t * NA + j] = e;
            s += e;
        }
        float inv = 1.0f / s;
        for (int j = 0; j < NA; ++j) Ss[t * NA + j] *= inv;
    }
    __syncthreads();

    // ---------------- Phase D: stream values, out[i] = sum_j w_ij*(v[j]+values_ij)
    // Thread map: d-quad = (t&15)*4 (16 threads cover DH -> 256B coalesced per
    // (i,j)); row group ig = t>>4 owns rows ig*8 .. ig*8+7.
    {
        const int d  = (t & 15) * 4;
        const int ig = t >> 4;
        const size_t jstride = (size_t)NH * DH;   // 512 floats between j's

        float4 acc[8];
        const float* bp[8];
#pragma unroll
        for (int s = 0; s < 8; ++s) {
            acc[s].x = acc[s].y = acc[s].z = acc[s].w = 0.0f;
            int i = ig * 8 + s;
            bp[s] = values + ((size_t)(c * (NA * NA) + i * NA) * NH + h) * DH + d;
        }

        for (int j = 0; j < NA; ++j) {
            float4 vj = *(const float4*)&Vs[j * DH + d];
#pragma unroll
            for (int s = 0; s < 8; ++s) {
                int i = ig * 8 + s;
                float p = Ss[i * NA + j];                      // LDS broadcast
                float4 e = *(const float4*)(bp[s] + (size_t)j * jstride);
                acc[s].x += p * (vj.x + e.x);
                acc[s].y += p * (vj.y + e.y);
                acc[s].z += p * (vj.z + e.z);
                acc[s].w += p * (vj.w + e.w);
            }
        }

#pragma unroll
        for (int s = 0; s < 8; ++s) {
            int i = ig * 8 + s;
            size_t o = ((size_t)((c * NA + i) * NH + h)) * DH + d;
            *(float4*)(out + o) = acc[s];
        }
    }
}

extern "C" void kernel_launch(void* const* d_in, const int* in_sizes, int n_in,
                              void* d_out, int out_size, void* d_ws, size_t ws_size,
                              hipStream_t stream) {
    (void)in_sizes; (void)n_in; (void)d_ws; (void)ws_size; (void)out_size;
    const float* q      = (const float*)d_in[0];
    const float* k      = (const float*)d_in[1];
    const float* v      = (const float*)d_in[2];
    const float* aw     = (const float*)d_in[3];
    const float* values = (const float*)d_in[4];
    // d_in[5] = edges (int32): analytically dense per-crystal blocks; unused.
    float* out = (float*)d_out;

    dim3 grid(G_CRY * NH);   // 512 blocks: one per (crystal, head)
    dim3 block(128);         // 4 wave32 waves
    crystal_attn_kernel<<<grid, block, 0, stream>>>(q, k, v, aw, values, out);
}
